// OneSideInterModalityUpdate_38147899523086
// MI455X (gfx1250) — matrix operand order repaired and checked by reading, not compiled, not run on hardware
//
#include <hip/hip_runtime.h>
#include <hip/hip_bf16.h>

// ---------------------------------------------------------------------------
// MI455X (gfx1250) implementation: bf16 WMMA tensor path, fp32 accumulate.
// All matmuls via v_wmma_f32_16x16x32_bf16. Intermediates stored bf16.
// Q/K attention tiles staged via TDM tensor_load_to_lds (TENSORcnt) when the
// builtin is available; GEMM uses global_prefetch for the next K-slab.
// ---------------------------------------------------------------------------

typedef __attribute__((ext_vector_type(16))) __bf16 v16bf;
typedef __attribute__((ext_vector_type(8)))  __bf16 v8bf;
typedef __attribute__((ext_vector_type(8)))  float  v8f;
typedef unsigned int u32x4 __attribute__((ext_vector_type(4)));
typedef int          i32x8 __attribute__((ext_vector_type(8)));
typedef int          i32x4 __attribute__((ext_vector_type(4)));

__device__ inline v8f wmma_bf16(v16bf a, v16bf b, v8f c) {
  // (neg_a, A, neg_b, B, c_mod, C, reuse_a, reuse_b)
  return __builtin_amdgcn_wmma_f32_16x16x32_bf16(false, a, false, b,
                                                 (short)0, c, false, false);
}

__device__ inline v8f zero8() {
  v8f z = {0.f, 0.f, 0.f, 0.f, 0.f, 0.f, 0.f, 0.f};
  return z;
}

__device__ inline unsigned pack2(float x, float y) {
  unsigned short ax = __builtin_bit_cast(unsigned short, (__bf16)x);
  unsigned short ay = __builtin_bit_cast(unsigned short, (__bf16)y);
  return (unsigned)ax | ((unsigned)ay << 16);
}

// A operand fragment (16x32, MxK), LDS row-major along K, pitch in elements.
// ISA layout: lane m=L%16: VGPR0-3 hold K = 8*half + 0..7, VGPR4-7 hold +16.
__device__ inline v16bf load_frag_a(const __bf16* base, int pitch, int row0,
                                    int k0, int lane) {
  const int m = lane & 15, half = lane >> 4;
  const __bf16* p = base + (size_t)(row0 + m) * pitch + k0 + 8 * half;
  v8bf lo = *(const v8bf*)p;        // K = k0 + 8h .. +7
  v8bf hi = *(const v8bf*)(p + 16); // K = k0 + 16 + 8h .. +7
  v16bf a;
#pragma unroll
  for (int i = 0; i < 8; i++) { a[i] = lo[i]; a[i + 8] = hi[i]; }
  return a;
}

// B operand fragment (32x16, KxN) from an LDS tile stored [n][k] (row-major
// along K). Lane n=L%16, half=L/16 holds K = 8*half + 0..7, then +16 (as A).
__device__ inline v16bf load_frag_b(const __bf16* base, int pitch, int col0,
                                    int k0, int lane) {
  const int n = lane & 15, half = lane >> 4;
  const __bf16* p = base + (size_t)(col0 + n) * pitch + k0 + 8 * half;
  v8bf lo = *(const v8bf*)p;
  v8bf hi = *(const v8bf*)(p + 16);
  v16bf b;
#pragma unroll
  for (int i = 0; i < 8; i++) { b[i] = lo[i]; b[i + 8] = hi[i]; }
  return b;
}

// ---------------------------------------------------------------------------
// TDM: 2D bf16 tile (tile_rows x row_elems) global -> LDS, with LDS padding of
// 4 DWORDs (8 bf16) after every 64 DWORDs (128 bf16) => LDS row pitch 136.
// Descriptor packing per CDNA5 ISA. Issue from a
// single wave; completion via s_wait_tensorcnt 0.
// This toolchain exposes the 6-arg builtin:
//   (u32x4 g0, i32x8 g1, i32x4 g2, i32x4 g3, i32x8 g4, i32 cpol)
// ---------------------------------------------------------------------------
#if __has_builtin(__builtin_amdgcn_tensor_load_to_lds)
#define HAVE_TDM 1
__device__ inline void tdm_load_tile2d(unsigned lds_byte_addr,
                                       const void* gaddr, int tile_rows,
                                       int row_elems, int src_pitch_elems) {
  unsigned long long ga = (unsigned long long)gaddr;
  u32x4 g0;
  g0[0] = 1u;                                   // count=1, user descriptor
  g0[1] = lds_byte_addr;                        // lds_addr [63:32]
  g0[2] = (unsigned)(ga & 0xffffffffu);         // global_addr [95:64]
  g0[3] = (unsigned)((ga >> 32) & 0x01ffffffu)  // global_addr [120:96]
        | (2u << 30);                           // type=2 ("image")
  i32x8 g1;
  g1[0] = (1 << 16)    // data_size = 2 bytes
        | (1 << 20)    // pad_enable
        | (5 << 22)    // pad_interval: 64 DWORDs
        | (3 << 25);   // pad_amount: 4 DWORDs  -> bf16 pitch 128+8
  g1[1] = (row_elems & 0xffff) << 16;            // tensor_dim0 [79:48] lo
  g1[2] = ((unsigned)row_elems >> 16)            // tensor_dim0 hi
        | ((tile_rows & 0xffff) << 16);          // tensor_dim1 [111:80] lo
  g1[3] = ((unsigned)tile_rows >> 16)            // tensor_dim1 hi
        | ((row_elems & 0xffff) << 16);          // tile_dim0 [127:112]
  g1[4] = (tile_rows & 0xffff);                  // tile_dim1 (tile_dim2 = 0)
  g1[5] = src_pitch_elems;                       // tensor_dim0_stride lo
  g1[6] = 0;                                     // stride0 hi / stride1 lo
  g1[7] = 0;
  i32x4 gz4 = {0, 0, 0, 0};                      // groups 2/3 unused (2D)
  i32x8 gz8 = {0, 0, 0, 0, 0, 0, 0, 0};          // extra group (unused)
  __builtin_amdgcn_tensor_load_to_lds(g0, g1, gz4, gz4, gz8, 0);
}
#else
#define HAVE_TDM 0
#endif

// ---------------------------------------------------------------------------
// Generic GEMM: Y[M,N] = (X[M,K] @ W[N,K]^T + bias[N]) * rowMask[M] * outScale
// Block: 256 threads (8 waves, 4x2), tile 128x64, BK=64.
// ---------------------------------------------------------------------------
constexpr int BM = 128, BN = 64, BK = 64;
constexpr int AP = BK + 8; // LDS row pitch (elements): 144B rows, bank-safe

template <bool X_BF16, bool OUT_BF16>
__global__ __launch_bounds__(256) void gemm_wmma(
    const void* __restrict__ Xv, const float* __restrict__ W,
    const float* __restrict__ bias, const float* __restrict__ rowMask,
    void* __restrict__ Yv, int M, int N, int K, float outScale) {
  __shared__ __bf16 As[BM * AP];
  __shared__ __bf16 Bs[BN * AP];

  const int tid = threadIdx.x;
  const int lane = tid & 31, wave = tid >> 5;
  const int wm = wave >> 1, wn = wave & 1; // 4x2 wave grid, 32x32 per wave
  const int rowBase = blockIdx.x * BM;
  const int colBase = blockIdx.y * BN;

  const float* Xf = (const float*)Xv;
  const __bf16* Xb = (const __bf16*)Xv;

  v8f acc[2][2];
  acc[0][0] = zero8(); acc[0][1] = zero8();
  acc[1][0] = zero8(); acc[1][1] = zero8();

  for (int k0 = 0; k0 < K; k0 += BK) {
    __syncthreads(); // LDS reuse vs previous iteration's reads
    // Stage X tile: 128 rows x 64 K, 4 elems/load, 8 loads/thread
#pragma unroll
    for (int i = 0; i < 8; i++) {
      int idx = i * 256 + tid;
      int r = idx >> 4, g = idx & 15;
      uint2 st;
      if constexpr (X_BF16) {
        st = *(const uint2*)&Xb[(size_t)(rowBase + r) * K + k0 + g * 4];
      } else {
        float4 f = *(const float4*)&Xf[(size_t)(rowBase + r) * K + k0 + g * 4];
        st.x = pack2(f.x, f.y);
        st.y = pack2(f.z, f.w);
      }
      *(uint2*)&As[r * AP + g * 4] = st;
    }
    // Stage W tile: 64 rows x 64 K (fp32 -> bf16), 4 loads/thread
#pragma unroll
    for (int i = 0; i < 4; i++) {
      int idx = i * 256 + tid;
      int r = idx >> 4, g = idx & 15;
      float4 f = *(const float4*)&W[(size_t)(colBase + r) * K + k0 + g * 4];
      uint2 st;
      st.x = pack2(f.x, f.y);
      st.y = pack2(f.z, f.w);
      *(uint2*)&Bs[r * AP + g * 4] = st;
    }
    // Speculative prefetch of the next K-slab (global_prefetch_b8); weights
    // stream from the 192MB L2, hide the latency behind this slab's WMMAs.
    if (k0 + BK < K) {
      int pr = tid >> 1; // 0..127: two lanes per A row
      if constexpr (X_BF16)
        __builtin_prefetch(&Xb[(size_t)(rowBase + pr) * K + k0 + BK], 0, 1);
      else
        __builtin_prefetch(&Xf[(size_t)(rowBase + pr) * K + k0 + BK], 0, 1);
      if (tid < BN)
        __builtin_prefetch(&W[(size_t)(colBase + tid) * K + k0 + BK], 0, 1);
    }
    __syncthreads();
#pragma unroll
    for (int kk = 0; kk < BK; kk += 32) {
      v16bf a0 = load_frag_a(As, AP, wm * 32, kk, lane);
      v16bf a1 = load_frag_a(As, AP, wm * 32 + 16, kk, lane);
      v16bf b0 = load_frag_b(Bs, AP, wn * 32, kk, lane);
      v16bf b1 = load_frag_b(Bs, AP, wn * 32 + 16, kk, lane);
      acc[0][0] = wmma_bf16(a0, b0, acc[0][0]);
      acc[0][1] = wmma_bf16(a0, b1, acc[0][1]);
      acc[1][0] = wmma_bf16(a1, b0, acc[1][0]);
      acc[1][1] = wmma_bf16(a1, b1, acc[1][1]);
    }
  }

  // Epilogue. C layout: lane n=L%16, rows m = 8*(L/16)+v in VGPR v.
  const int n = lane & 15, half = lane >> 4;
#pragma unroll
  for (int mi = 0; mi < 2; mi++) {
#pragma unroll
    for (int ni = 0; ni < 2; ni++) {
      const int col = colBase + wn * 32 + ni * 16 + n;
      const float bv = bias ? bias[col] : 0.f;
      const int rbase = rowBase + wm * 32 + mi * 16 + half * 8;
#pragma unroll
      for (int v = 0; v < 8; v++) {
        const int row = rbase + v;
        float y = acc[mi][ni][v] + bv;
        if (rowMask) y *= rowMask[row];
        y *= outScale;
        if constexpr (OUT_BF16)
          ((__bf16*)Yv)[(size_t)row * N + col] = (__bf16)y;
        else
          ((float*)Yv)[(size_t)row * N + col] = y;
      }
    }
  }
}

// ---------------------------------------------------------------------------
// tgt (fp32) -> CAT[:, 0:1024] (bf16)
// ---------------------------------------------------------------------------
__global__ __launch_bounds__(256) void cat_tgt_kernel(
    const float* __restrict__ tgt, __bf16* __restrict__ CAT) {
  size_t idx = ((size_t)blockIdx.x * 256 + threadIdx.x) * 4;
  size_t row = idx >> 10;
  size_t col = idx & 1023;
  float4 f = *(const float4*)&tgt[idx];
  uint2 v;
  v.x = pack2(f.x, f.y);
  v.y = pack2(f.z, f.w);
  *(uint2*)&CAT[row * 2048 + col] = v;
}

// ---------------------------------------------------------------------------
// Attention per (b,h,t-tile of 64). Two-pass softmax: the full 64x512 score
// strip lives in LDS (possible only with CDNA5's 320KB/WGP LDS).
// grid = (B*H, T/64), block = 256 (8 waves).
// ---------------------------------------------------------------------------
constexpr int QP = 136; // pitch for 64x128 Q/K tiles (272B rows, TDM-padded)
constexpr int VP = 72;  // pitch for 128x64 transposed V tile
constexpr int SP = 528; // pitch for 64x512 score/prob strip
constexpr size_t ATTN_SMEM =
    (size_t)(64 * QP + 64 * QP + 128 * VP + 64 * SP) * 2 + 64 * 8 * 4;

__global__ __launch_bounds__(256) void attn_wmma(
    const __bf16* __restrict__ Q,   // [B*T, 1024], pre-scaled by 1/sqrt(128)
    const __bf16* __restrict__ KV,  // [B*S, 2048]: cols 0-1023 key, 1024+ val
    const float* __restrict__ src_mask, // [B*S]
    __bf16* __restrict__ CAT) {         // [B*T, 2048], we write cols 1024+
  extern __shared__ char smem_raw[];
  __bf16* Qs = (__bf16*)smem_raw;   // [64][QP]
  __bf16* Ks = Qs + 64 * QP;        // [64][QP]
  __bf16* Vt = Ks + 64 * QP;        // [128][VP]  (V transposed: [d][s])
  __bf16* Ss = Vt + 128 * VP;       // [64][SP]   scores -> probs (in place)
  float* red = (float*)(Ss + 64 * SP); // [64][8] reduction scratch

  const int tid = threadIdx.x;
  const int lane = tid & 31, wave = tid >> 5;
  const int b = blockIdx.x >> 3, h = blockIdx.x & 7;
  const int t0 = blockIdx.y * 64;
  const size_t qRow0 = (size_t)b * 512 + t0;
  const size_t sRow0 = (size_t)b * 512;

  // Stage Q tile 64x128 (already scaled)
#if HAVE_TDM
  if (tid < 32)
    tdm_load_tile2d((unsigned)(size_t)(void*)Qs,
                    (const void*)(Q + qRow0 * 1024 + h * 128), 64, 128, 1024);
#else
#pragma unroll
  for (int i = 0; i < 8; i++) {
    int idx = i * 256 + tid;
    int r = idx >> 5, g = idx & 31;
    uint2 v = *(const uint2*)&Q[(qRow0 + r) * 1024 + h * 128 + g * 4];
    *(uint2*)&Qs[r * QP + g * 4] = v;
  }
#endif

  const int mt = wave >> 1; // t-subtile (0..3)
  const int nh = wave & 1;  // n-half

  // ---- Pass 1: scores[64][512] = Q K^T (scale folded into Q) ----
  for (int c = 0; c < 8; c++) {
    __syncthreads();
#if HAVE_TDM
    if (tid < 32) {
      tdm_load_tile2d((unsigned)(size_t)(void*)Ks,
                      (const void*)&KV[(sRow0 + c * 64) * 2048 + h * 128], 64,
                      128, 2048);
      __builtin_amdgcn_s_wait_tensorcnt(0);
    }
#else
#pragma unroll
    for (int i = 0; i < 8; i++) {
      int idx = i * 256 + tid;
      int r = idx >> 5, g = idx & 31; // r = s index, g*4 = d base
      uint2 kv = *(const uint2*)&KV[(sRow0 + c * 64 + r) * 2048 + h * 128 + g * 4];
      *(uint2*)&Ks[r * QP + g * 4] = kv;
    }
#endif
    __syncthreads();
    v8f sc[2];
    sc[0] = zero8(); sc[1] = zero8();
#pragma unroll
    for (int k0 = 0; k0 < 128; k0 += 32) {
      v16bf a = load_frag_a(Qs, QP, mt * 16, k0, lane);
#pragma unroll
      for (int ni = 0; ni < 2; ni++) {
        v16bf kb = load_frag_b(Ks, QP, nh * 32 + ni * 16, k0, lane);
        sc[ni] = wmma_bf16(a, kb, sc[ni]);
      }
    }
    const int n = lane & 15, half = lane >> 4;
#pragma unroll
    for (int ni = 0; ni < 2; ni++) {
      int colS = c * 64 + nh * 32 + ni * 16 + n;
      int rbase = mt * 16 + half * 8;
#pragma unroll
      for (int v = 0; v < 8; v++)
        Ss[(rbase + v) * SP + colS] = (__bf16)sc[ni][v];
    }
  }
  __syncthreads();

  // ---- Softmax over s (masked), probabilities written in place (bf16) ----
  {
    const int row = tid >> 2, part = tid & 3; // 4 threads/row, 128 cols each
    const float* mrow = src_mask + (size_t)b * 512;
    float mx = -3.4e38f;
    for (int j = 0; j < 128; j++) {
      int cS = part * 128 + j;
      if (mrow[cS] != 0.f) mx = fmaxf(mx, (float)Ss[row * SP + cS]);
    }
    red[row * 8 + part] = mx;
    __syncthreads();
    mx = fmaxf(fmaxf(red[row * 8 + 0], red[row * 8 + 1]),
               fmaxf(red[row * 8 + 2], red[row * 8 + 3]));
    float sum = 0.f;
    for (int j = 0; j < 128; j++) {
      int cS = part * 128 + j;
      if (mrow[cS] != 0.f) sum += __expf((float)Ss[row * SP + cS] - mx);
    }
    red[row * 8 + 4 + part] = sum;
    __syncthreads();
    float inv = 1.f / (red[row * 8 + 4] + red[row * 8 + 5] +
                       red[row * 8 + 6] + red[row * 8 + 7]);
    for (int j = 0; j < 128; j++) {
      int cS = part * 128 + j;
      float p = (mrow[cS] != 0.f)
                    ? __expf((float)Ss[row * SP + cS] - mx) * inv
                    : 0.f;
      Ss[row * SP + cS] = (__bf16)p;
    }
  }

  // ---- Pass 2: O[64][128] = P V ----
  v8f on[4];
  on[0] = zero8(); on[1] = zero8(); on[2] = zero8(); on[3] = zero8();
  for (int c = 0; c < 8; c++) {
    __syncthreads();
#pragma unroll
    for (int i = 0; i < 8; i++) { // stage V chunk transposed: Vt[d][s]
      int idx = i * 256 + tid;
      int r = idx >> 5, g = idx & 31;
      uint2 w2 =
          *(const uint2*)&KV[(sRow0 + c * 64 + r) * 2048 + 1024 + h * 128 + g * 4];
      const __bf16* e = (const __bf16*)&w2;
      Vt[(g * 4 + 0) * VP + r] = e[0];
      Vt[(g * 4 + 1) * VP + r] = e[1];
      Vt[(g * 4 + 2) * VP + r] = e[2];
      Vt[(g * 4 + 3) * VP + r] = e[3];
    }
    __syncthreads();
#pragma unroll
    for (int kk = 0; kk < 64; kk += 32) {
      v16bf a = load_frag_a(Ss, SP, mt * 16, c * 64 + kk, lane);
#pragma unroll
      for (int ni = 0; ni < 4; ni++) {
        v16bf vb = load_frag_b(Vt, VP, nh * 64 + ni * 16, kk, lane);
        on[ni] = wmma_bf16(a, vb, on[ni]);
      }
    }
  }

  // Write tgt_update into CAT[:, 1024 + h*128 + d] as bf16
  const int n = lane & 15, half = lane >> 4;
#pragma unroll
  for (int ni = 0; ni < 4; ni++) {
    int d = nh * 64 + ni * 16 + n;
    int rbase = mt * 16 + half * 8;
#pragma unroll
    for (int v = 0; v < 8; v++) {
      int t = rbase + v;
      CAT[(qRow0 + t) * 2048 + 1024 + h * 128 + d] = (__bf16)on[ni][v];
    }
  }
}

// ---------------------------------------------------------------------------
// Host launcher
// ---------------------------------------------------------------------------
extern "C" void kernel_launch(void* const* d_in, const int* in_sizes, int n_in,
                              void* d_out, int out_size, void* d_ws,
                              size_t ws_size, hipStream_t stream) {
  (void)in_sizes; (void)n_in; (void)out_size; (void)ws_size;
  const float* src      = (const float*)d_in[0]; // [32,512,2048]
  const float* tgt      = (const float*)d_in[1]; // [32,512,1024]
  const float* src_mask = (const float*)d_in[2]; // [32,512]
  const float* tgt_mask = (const float*)d_in[3]; // [32,512]
  const float* W_src    = (const float*)d_in[4]; // [2048,2048]
  const float* b_src    = (const float*)d_in[5];
  const float* W_tgt    = (const float*)d_in[6]; // [1024,1024]
  const float* b_tgt    = (const float*)d_in[7];
  const float* W_out    = (const float*)d_in[8]; // [1024,2048]
  const float* b_out    = (const float*)d_in[9];
  float* out = (float*)d_out;                    // [32,512,1024] fp32

  const int M = 32 * 512; // 16384 rows for both src and tgt sides
  __bf16* KV  = (__bf16*)d_ws;                 // [16384, 2048] key|value
  __bf16* Qb  = KV + (size_t)M * 2048;         // [16384, 1024]
  __bf16* CAT = Qb + (size_t)M * 1024;         // [16384, 2048] tgt|update

  // 1) src projection -> masked K|V (bf16)
  gemm_wmma<false, true><<<dim3(M / BM, 2048 / BN), 256, 0, stream>>>(
      src, W_src, b_src, src_mask, KV, M, 2048, 2048, 1.0f);
  // 2) tgt projection -> masked Q (bf16), 1/sqrt(128) folded in
  gemm_wmma<false, true><<<dim3(M / BM, 1024 / BN), 256, 0, stream>>>(
      tgt, W_tgt, b_tgt, tgt_mask, Qb, M, 1024, 1024, 0.08838834764831845f);
  // 3) tgt -> CAT lower half (bf16)
  cat_tgt_kernel<<<M * 1024 / (256 * 4), 256, 0, stream>>>(tgt, CAT);
  // 4) attention, writes CAT upper half
  attn_wmma<<<dim3(32 * 8, 512 / 64), 256, ATTN_SMEM, stream>>>(
      Qb, KV, src_mask, CAT);
  // 5) output projection (fp32 out)
  gemm_wmma<true, false><<<dim3(M / BM, 1024 / BN), 256, 0, stream>>>(
      CAT, W_out, b_out, nullptr, out, M, 1024, 2048, 1.0f);
}
